// GPT_15504831939091
// MI455X (gfx1250) — matrix operand order repaired and checked by reading, not compiled, not run on hardware
//
#include <hip/hip_runtime.h>
#include <hip/hip_bf16.h>
#include <math.h>

typedef __bf16 bf16_t;
typedef __attribute__((ext_vector_type(16))) __bf16 v16bf;
typedef __attribute__((ext_vector_type(8)))  __bf16 v8bf;
typedef __attribute__((ext_vector_type(8)))  float  v8f;

#define NHEAD  12
#define DMODEL 768
#define DHEAD  64
#define SEQ    1024
#define NBATCH 4
#define NROWS  (NBATCH * SEQ)   // 4096
#define FFDIM  3072
#define VOCAB  50257
#define NLAYER 12

union F8 { v8f v; float f[8]; };

// ---- WMMA fragment helpers (wave32; contiguous k order, same in A and B) ----------
// A 16x32 bf16: lane holds row m=lane&15; K = (lane>>4)*16 .. +15 contiguous
// (same order as B).  -> two 16B loads from a row-major row.
__device__ __forceinline__ v16bf load_a_frag(const bf16_t* p) {
  union { v16bf v; v8bf h[2]; } u;
  u.h[0] = *(const v8bf*)(p);
  u.h[1] = *(const v8bf*)(p + 8);
  return u.v;
}
// B 32x16 bf16: lane holds col n=lane&15; K = (lane>>4)*16 .. +15 contiguous.
__device__ __forceinline__ v16bf load_b_frag(const bf16_t* p) {
  union { v16bf v; v8bf h[2]; } u;
  u.h[0] = *(const v8bf*)(p);
  u.h[1] = *(const v8bf*)(p + 8);
  return u.v;
}
__device__ __forceinline__ v8f wmma_bf16(v16bf a, v16bf b, v8f c) {
  return __builtin_amdgcn_wmma_f32_16x16x32_bf16(false, a, false, b, (short)0, c,
                                                 false, false);
}

// ---- CDNA5 async copy: global -> LDS, 16B per lane, tracked by ASYNCcnt ----
__device__ __forceinline__ void async_copy_b128(unsigned lds_off,
                                                const void* gaddr) {
  asm volatile("global_load_async_to_lds_b128 %0, %1, off"
               :: "v"(lds_off), "v"((unsigned long long)gaddr)
               : "memory");
}
template <int N>
__device__ __forceinline__ void wait_asynccnt() {
  if constexpr (N == 0) asm volatile("s_wait_asynccnt 0x0" ::: "memory");
  else                  asm volatile("s_wait_asynccnt 0x1" ::: "memory");
}

// ---- embedding: x[b,s,:] = wte[idx[b,s],:] + wpe[s,:] ----------------------
__global__ __launch_bounds__(256) void embed_kernel(
    const int* __restrict__ idx, const float* __restrict__ wte,
    const float* __restrict__ wpe, float* __restrict__ x)
{
  const int row = blockIdx.x;          // b*S + s
  const int s = row % SEQ;
  const int tok = idx[row];
  for (int i = threadIdx.x; i < DMODEL; i += 256)
    x[(size_t)row * DMODEL + i] =
        wte[(size_t)tok * DMODEL + i] + wpe[(size_t)s * DMODEL + i];
}

// ---- layernorm fp32 row -> bf16 row ----------------------------------------
__global__ __launch_bounds__(256) void layernorm_kernel(
    const float* __restrict__ x, const float* __restrict__ g,
    const float* __restrict__ be, bf16_t* __restrict__ out)
{
  const int row = blockIdx.x;
  const int tid = threadIdx.x;
  const float* xr = x + (size_t)row * DMODEL;
  float s1 = 0.f, s2 = 0.f;
  for (int i = tid; i < DMODEL; i += 256) { float v = xr[i]; s1 += v; s2 += v * v; }
#pragma unroll
  for (int m = 1; m < 32; m <<= 1) {
    s1 += __shfl_xor(s1, m, 32);
    s2 += __shfl_xor(s2, m, 32);
  }
  __shared__ float a1[8], a2[8];
  if ((tid & 31) == 0) { a1[tid >> 5] = s1; a2[tid >> 5] = s2; }
  __syncthreads();
  float t1 = 0.f, t2 = 0.f;
#pragma unroll
  for (int i = 0; i < 8; ++i) { t1 += a1[i]; t2 += a2[i]; }
  const float mean = t1 / (float)DMODEL;
  const float var = t2 / (float)DMODEL - mean * mean;
  const float rs = rsqrtf(var + 1e-5f);
  bf16_t* orr = out + (size_t)row * DMODEL;
  for (int i = tid; i < DMODEL; i += 256)
    orr[i] = (bf16_t)((xr[i] - mean) * rs * g[i] + be[i]);
}

// ---- W [K,N] fp32 -> Wt [N,K] bf16 -----------------------------------------
__global__ __launch_bounds__(256) void convt_kernel(
    const float* __restrict__ W, bf16_t* __restrict__ Wt, int K, int N)
{
  size_t i = (size_t)blockIdx.x * 256 + threadIdx.x;
  if (i >= (size_t)K * N) return;
  int kk = (int)(i / N), n = (int)(i % N);
  Wt[(size_t)n * K + kk] = (bf16_t)W[i];
}

// ---- V [B,S,D] bf16 -> Vt [B,H,Dh,S] bf16 ----------------------------------
__global__ __launch_bounds__(256) void vtrans_kernel(
    const bf16_t* __restrict__ v, bf16_t* __restrict__ vt)
{
  size_t i = (size_t)blockIdx.x * 256 + threadIdx.x;
  if (i >= (size_t)NROWS * DMODEL) return;
  int d = (int)(i % DMODEL);
  int row = (int)(i / DMODEL);
  int b = row / SEQ, s = row % SEQ;
  int h = d / DHEAD, dh = d % DHEAD;
  vt[((size_t)(b * NHEAD + h) * DHEAD + dh) * SEQ + s] = v[i];
}

// ---- bf16 GEMM: out[M,N] = A[M,K] @ Bt[N,K]^T, f32 accumulate --------------
// Block: 256 thr (8 waves); block tile 256(M) x 64(N); wave tile 32x64.
// B tile (64x32 bf16) staged in LDS via async global->LDS copies (ASYNCcnt),
// double-buffered; LDS rows padded to 40 bf16 for conflict-free ds_load_b128.
// EPI 0: +bias -> bf16 ; 1: gelu(+bias) -> bf16 ; 2: +bias+residual -> f32 ;
// EPI 3: plain -> f32 (lm head, ragged N).
#define BROW 40   // padded LDS row length in bf16 elements (80 B)
template <int EPI>
__global__ __launch_bounds__(256) void gemm_wmma_kernel(
    const bf16_t* __restrict__ A, const bf16_t* __restrict__ Bt,
    const float* __restrict__ bias, const float* resid, void* outp,
    int M, int N, int K)
{
  __shared__ bf16_t bsh[2][64 * BROW];

  const int tid  = threadIdx.x;
  const int lane = tid & 31;
  const int wave = tid >> 5;
  const int half = lane >> 4;
  const int l15  = lane & 15;
  const int m0 = blockIdx.y * 256 + wave * 32;
  const int n0 = blockIdx.x * 64;

  // per-thread async staging task: one 16B chunk of the 64x32 B tile
  const int srow = tid >> 2;          // 0..63 : tile row (n within tile)
  const int sc4  = tid & 3;           // 0..3  : 16B chunk within 32-elem row
  int nsrc = n0 + srow;
  if (nsrc >= N) nsrc = N - 1;        // clamp (lm head); stores are guarded
  const bf16_t* bsrc = Bt + (size_t)nsrc * K + sc4 * 8;
  const unsigned lds0 =
      (unsigned)(unsigned long long)(const void*)&bsh[0][srow * BROW + sc4 * 8];
  const unsigned lds1 =
      (unsigned)(unsigned long long)(const void*)&bsh[1][srow * BROW + sc4 * 8];

  F8 acc[8] = {};
  const bf16_t* arow0 = A + (size_t)(m0 + l15) * K + half * 16;
  const bf16_t* arow1 = arow0 + (size_t)16 * K;

  const int nsteps = K / 32;
  async_copy_b128(lds0, bsrc);                      // prologue: k = 0 -> buf 0
  for (int i = 0; i < nsteps; ++i) {
    const int k = i * 32;
    const bool pre = (i + 1) < nsteps;
    if (pre) {
      async_copy_b128((i & 1) ? lds0 : lds1, bsrc + k + 32);
      __builtin_prefetch(arow0 + k + 32, 0, 1);     // global_prefetch next A
      __builtin_prefetch(arow1 + k + 32, 0, 1);
      wait_asynccnt<1>();                            // current buffer ready
    } else {
      wait_asynccnt<0>();
    }
    __syncthreads();                                 // tile visible to block
    const bf16_t* bbuf = bsh[i & 1];
    const v16bf a0 = load_a_frag(arow0 + k);
    const v16bf a1 = load_a_frag(arow1 + k);
#pragma unroll
    for (int t = 0; t < 4; ++t) {
      const v16bf b = load_b_frag(&bbuf[(t * 16 + l15) * BROW + half * 16]);
      acc[t].v     = wmma_bf16(a0, b, acc[t].v);
      acc[t + 4].v = wmma_bf16(a1, b, acc[t + 4].v);
    }
    __syncthreads();                                 // done reading this buf
  }

#pragma unroll
  for (int mh = 0; mh < 2; ++mh) {
#pragma unroll
    for (int t = 0; t < 4; ++t) {
      const int n = n0 + t * 16 + l15;
      if (n >= N) continue;
      float bv = 0.f;
      if (EPI != 3) bv = bias[n];
#pragma unroll
      for (int r = 0; r < 8; ++r) {
        const int m = m0 + mh * 16 + half * 8 + r;
        const size_t o = (size_t)m * N + n;
        float v = acc[mh * 4 + t].f[r] + bv;
        if (EPI == 1) v = 0.5f * v * (1.0f + erff(v * 0.70710678118654752f));
        if (EPI == 2) v += resid[o];
        if (EPI == 0 || EPI == 1) ((bf16_t*)outp)[o] = (bf16_t)v;
        else                      ((float*)outp)[o]  = v;
      }
    }
  }
}

// ---- flash attention: one wave per (b, h, 16-query tile) -------------------
__global__ __launch_bounds__(32) void attn_kernel(
    const bf16_t* __restrict__ q, const bf16_t* __restrict__ k,
    const bf16_t* __restrict__ vt, bf16_t* __restrict__ o)
{
  const int lane = threadIdx.x & 31;
  const int half = lane >> 4;
  const int l15  = lane & 15;
  const int QT = SEQ / 16;
  const int qt = blockIdx.x % QT;
  const int h  = (blockIdx.x / QT) % NHEAD;
  const int b  = blockIdx.x / (QT * NHEAD);
  const int q0 = qt * 16;

  __shared__ bf16_t pls[16 * 32];   // P tile staging (C-layout -> A-layout)

  const bf16_t* qrow =
      q + (size_t)(b * SEQ + q0 + l15) * DMODEL + h * DHEAD + half * 16;
  const v16bf qa0 = load_a_frag(qrow);        // dh 0..31
  const v16bf qa1 = load_a_frag(qrow + 32);   // dh 32..63

  F8 oacc[4] = {};
  float rowmax[8], rowsum[8];
#pragma unroll
  for (int r = 0; r < 8; ++r) { rowmax[r] = -1e30f; rowsum[r] = 0.f; }

  for (int kb0 = 0; kb0 <= q0 + 15; kb0 += 32) {
    F8 s[2] = {};
#pragma unroll
    for (int sub = 0; sub < 2; ++sub) {
      int krow = kb0 + sub * 16 + l15;
      if (krow > SEQ - 1) krow = SEQ - 1;   // masked below anyway
      const bf16_t* kr =
          k + (size_t)(b * SEQ + krow) * DMODEL + h * DHEAD + half * 16;
      s[sub].v = wmma_bf16(qa0, load_b_frag(kr),      s[sub].v);
      s[sub].v = wmma_bf16(qa1, load_b_frag(kr + 32), s[sub].v);
    }
    // scale 1/sqrt(64), causal mask, online softmax (rows live in lane halves)
#pragma unroll
    for (int r = 0; r < 8; ++r) {
      const int qm = q0 + half * 8 + r;
      float v0 = s[0].f[r] * 0.125f;
      float v1 = s[1].f[r] * 0.125f;
      if (kb0 + l15 > qm)      v0 = -1e30f;
      if (kb0 + 16 + l15 > qm) v1 = -1e30f;
      float tm = fmaxf(v0, v1);
#pragma unroll
      for (int mm = 1; mm < 16; mm <<= 1) tm = fmaxf(tm, __shfl_xor(tm, mm, 32));
      const float mnew  = fmaxf(rowmax[r], tm);
      const float alpha = __expf(rowmax[r] - mnew);
      rowmax[r] = mnew;
      const float p0 = __expf(v0 - mnew);
      const float p1 = __expf(v1 - mnew);
      float ps = p0 + p1;
#pragma unroll
      for (int mm = 1; mm < 16; mm <<= 1) ps += __shfl_xor(ps, mm, 32);
      rowsum[r] = rowsum[r] * alpha + ps;
#pragma unroll
      for (int t = 0; t < 4; ++t) oacc[t].f[r] *= alpha;
      pls[(half * 8 + r) * 32 + l15]      = (bf16_t)p0;
      pls[(half * 8 + r) * 32 + 16 + l15] = (bf16_t)p1;
    }
    __syncthreads();
    const v16bf pa = load_a_frag(&pls[l15 * 32 + half * 16]);
    __syncthreads();
#pragma unroll
    for (int t = 0; t < 4; ++t) {
      const bf16_t* vr = vt +
          ((size_t)(b * NHEAD + h) * DHEAD + t * 16 + l15) * SEQ +
          kb0 + half * 16;
      oacc[t].v = wmma_bf16(pa, load_b_frag(vr), oacc[t].v);
    }
  }
#pragma unroll
  for (int t = 0; t < 4; ++t)
#pragma unroll
    for (int r = 0; r < 8; ++r) {
      const size_t row = (size_t)(b * SEQ + q0 + half * 8 + r);
      o[row * DMODEL + h * DHEAD + t * 16 + l15] =
          (bf16_t)(oacc[t].f[r] / rowsum[r]);
    }
}

// ---------------------------------------------------------------------------
extern "C" void kernel_launch(void* const* d_in, const int* in_sizes, int n_in,
                              void* d_out, int out_size, void* d_ws, size_t ws_size,
                              hipStream_t stream)
{
  (void)in_sizes; (void)n_in; (void)out_size; (void)ws_size;
  const int*   idx   = (const int*)  d_in[0];
  const float* wte   = (const float*)d_in[1];
  const float* wpe   = (const float*)d_in[2];
  const float* ln1_g = (const float*)d_in[3];
  const float* ln1_b = (const float*)d_in[4];
  const float* wq    = (const float*)d_in[5];
  const float* bq    = (const float*)d_in[6];
  const float* wk    = (const float*)d_in[7];
  const float* bk    = (const float*)d_in[8];
  const float* wv    = (const float*)d_in[9];
  const float* bv    = (const float*)d_in[10];
  const float* wo    = (const float*)d_in[11];
  const float* bo    = (const float*)d_in[12];
  const float* ln2_g = (const float*)d_in[13];
  const float* ln2_b = (const float*)d_in[14];
  const float* w1    = (const float*)d_in[15];
  const float* b1    = (const float*)d_in[16];
  const float* w2    = (const float*)d_in[17];
  const float* b2    = (const float*)d_in[18];
  const float* lnf_g = (const float*)d_in[19];
  const float* lnf_b = (const float*)d_in[20];
  const float* lm_w  = (const float*)d_in[21];
  float* logits = (float*)d_out;

  char* base = (char*)d_ws;
  size_t off = 0;
  auto take = [&](size_t bytes) -> void* {
    void* p = base + off;
    off = (off + bytes + 255) & ~(size_t)255;
    return p;
  };
  float*  x    = (float*) take((size_t)NROWS * DMODEL * 4);
  bf16_t* xb   = (bf16_t*)take((size_t)NROWS * DMODEL * 2);
  bf16_t* qb   = (bf16_t*)take((size_t)NROWS * DMODEL * 2);
  bf16_t* kbuf = (bf16_t*)take((size_t)NROWS * DMODEL * 2);
  bf16_t* vb   = (bf16_t*)take((size_t)NROWS * DMODEL * 2);
  bf16_t* vtb  = (bf16_t*)take((size_t)NROWS * DMODEL * 2 + 256);
  bf16_t* ob   = (bf16_t*)take((size_t)NROWS * DMODEL * 2);
  bf16_t* ffb  = (bf16_t*)take((size_t)NROWS * FFDIM * 2);
  bf16_t* wqt  = (bf16_t*)take((size_t)DMODEL * DMODEL * 2);
  bf16_t* wkt  = (bf16_t*)take((size_t)DMODEL * DMODEL * 2);
  bf16_t* wvt  = (bf16_t*)take((size_t)DMODEL * DMODEL * 2);
  bf16_t* wot  = (bf16_t*)take((size_t)DMODEL * DMODEL * 2);
  bf16_t* w1t  = (bf16_t*)take((size_t)DMODEL * FFDIM * 2);
  bf16_t* w2t  = (bf16_t*)take((size_t)FFDIM * DMODEL * 2);
  bf16_t* lmt  = (bf16_t*)take((size_t)DMODEL * VOCAB * 2);

  const dim3 blk(256);
  embed_kernel<<<NROWS, blk, 0, stream>>>(idx, wte, wpe, x);

  auto convt = [&](const float* W, bf16_t* Wt, int K, int N) {
    size_t tot = (size_t)K * N;
    convt_kernel<<<(unsigned)((tot + 255) / 256), blk, 0, stream>>>(W, Wt, K, N);
  };

  const dim3 g768((DMODEL + 63) / 64, NROWS / 256);
  const dim3 gff((FFDIM + 63) / 64,  NROWS / 256);
  const dim3 glm((VOCAB + 63) / 64,  NROWS / 256);

  for (int l = 0; l < NLAYER; ++l) {
    const size_t wD = (size_t)l * DMODEL * DMODEL;
    const size_t wF = (size_t)l * DMODEL * FFDIM;
    const size_t vD = (size_t)l * DMODEL;
    const size_t vF = (size_t)l * FFDIM;

    layernorm_kernel<<<NROWS, blk, 0, stream>>>(x, ln1_g + vD, ln1_b + vD, xb);
    convt(wq + wD, wqt, DMODEL, DMODEL);
    convt(wk + wD, wkt, DMODEL, DMODEL);
    convt(wv + wD, wvt, DMODEL, DMODEL);
    gemm_wmma_kernel<0><<<g768, blk, 0, stream>>>(xb, wqt, bq + vD, nullptr, qb,
                                                  NROWS, DMODEL, DMODEL);
    gemm_wmma_kernel<0><<<g768, blk, 0, stream>>>(xb, wkt, bk + vD, nullptr, kbuf,
                                                  NROWS, DMODEL, DMODEL);
    gemm_wmma_kernel<0><<<g768, blk, 0, stream>>>(xb, wvt, bv + vD, nullptr, vb,
                                                  NROWS, DMODEL, DMODEL);
    vtrans_kernel<<<(unsigned)(((size_t)NROWS * DMODEL + 255) / 256), blk, 0,
                    stream>>>(vb, vtb);
    attn_kernel<<<NBATCH * NHEAD * (SEQ / 16), dim3(32), 0, stream>>>(qb, kbuf,
                                                                      vtb, ob);
    convt(wo + wD, wot, DMODEL, DMODEL);
    gemm_wmma_kernel<2><<<g768, blk, 0, stream>>>(ob, wot, bo + vD, x, x,
                                                  NROWS, DMODEL, DMODEL);
    layernorm_kernel<<<NROWS, blk, 0, stream>>>(x, ln2_g + vD, ln2_b + vD, xb);
    convt(w1 + wF, w1t, DMODEL, FFDIM);
    gemm_wmma_kernel<1><<<gff, blk, 0, stream>>>(xb, w1t, b1 + vF, nullptr, ffb,
                                                 NROWS, FFDIM, DMODEL);
    convt(w2 + wF, w2t, FFDIM, DMODEL);
    gemm_wmma_kernel<2><<<g768, blk, 0, stream>>>(ffb, w2t, b2 + vD, x, x,
                                                  NROWS, DMODEL, FFDIM);
  }

  layernorm_kernel<<<NROWS, blk, 0, stream>>>(x, lnf_g, lnf_b, xb);
  convt(lm_w, lmt, DMODEL, VOCAB);
  gemm_wmma_kernel<3><<<glm, blk, 0, stream>>>(xb, lmt, nullptr, nullptr, logits,
                                               NROWS, VOCAB, DMODEL);
}
